// TransBlock_29927332118745
// MI455X (gfx1250) — matrix operand order repaired and checked by reading, not compiled, not run on hardware
//
#include <hip/hip_runtime.h>
#include <math.h>

// ---------------------------------------------------------------------------
// Hypergraph transformer block for MI455X (gfx1250, wave32, WMMA).
//   N_NODE=50000, N_HE=25000, E=300000, D=256, H=8, DH=32, D_MID=512
// All dense GEMMs run on v_wmma_f32_16x16x32_bf16 (bf16 in, fp32 accum).
// Edge gathers hit L2-resident projected tables (192MB L2 >> 77MB tables).
// ---------------------------------------------------------------------------

typedef __attribute__((ext_vector_type(16))) __bf16          v16bf;
typedef __attribute__((ext_vector_type(8)))  float           v8f;
typedef __attribute__((ext_vector_type(8)))  unsigned int    v8u;

#define D_MODEL 256
#define D_MID   512
#define N_HEADS 8

// ---- helpers ---------------------------------------------------------------

__device__ inline unsigned short f2bf(float x) {
  unsigned u = __float_as_uint(x);
  u += 0x7FFFu + ((u >> 16) & 1u);           // round-to-nearest-even
  return (unsigned short)(u >> 16);
}
__device__ inline float bf2f(unsigned short b) {
  return __uint_as_float(((unsigned)b) << 16);
}
__device__ inline float gelu_tanh(float x) {
  float x3 = x * x * x;
  return 0.5f * x * (1.0f + tanhf(0.7978845608028654f * (x + 0.044715f * x3)));
}
__device__ inline void atomicMaxF(float* addr, float val) {
  if (val >= 0.0f) atomicMax((int*)addr, __float_as_int(val));
  else             atomicMin((unsigned int*)addr, __float_as_uint(val));
}

// A fragment: 16x32 bf16, lane L holds row M=L&15; K layout per ISA 7.12.2.
__device__ inline v16bf load_a_frag(const unsigned short* lds, int m0, int k0, int ldk) {
  int lane = threadIdx.x & 31;
  const unsigned short* row = lds + (m0 + (lane & 15)) * ldk + k0 + ((lane >> 4) << 3);
  v8u t;
#pragma unroll
  for (int p = 0; p < 8; ++p) {
    int ks = 2 * p + ((p & 4) ? 8 : 0);      // pairs: K = {0..7,16..23}(+8 for hi lanes)
    t[p] = *(const unsigned int*)(row + ks);
  }
  return __builtin_bit_cast(v16bf, t);
}
// B fragment: 32x16 bf16 from LDS stored [n][k]; K layout matches A: {0..7,16..23}(+8 for hi lanes).
__device__ inline v16bf load_b_frag(const unsigned short* lds, int n0, int k0, int ldk) {
  int lane = threadIdx.x & 31;
  const unsigned short* row = lds + (n0 + (lane & 15)) * ldk + k0 + ((lane >> 4) << 3);
  v8u t;
#pragma unroll
  for (int p = 0; p < 8; ++p) t[p] = *(const unsigned int*)(row + 2 * p + ((p & 4) ? 8 : 0));
  return __builtin_bit_cast(v16bf, t);
}

// ---- generic WMMA GEMM: C = ep(A@W + bias) (+R) ----------------------------
// A[M,K], W[K,N], C[M,N] fp32 row-major. ep bit0: gelu, bit1: add R.
// Block: 256 threads (8 waves), tile 128(M) x 64(N), K step 32.

__global__ __launch_bounds__(256) void gemm_kernel(
    const float* __restrict__ A, const float* __restrict__ W,
    const float* __restrict__ bias, const float* __restrict__ R,
    float* __restrict__ C, int M, int N, int K, int ep)
{
  __shared__ unsigned short lA[128 * 32];   // [m][k]
  __shared__ unsigned short lB[64 * 32];    // [n][k]
  int tid = threadIdx.x, wave = tid >> 5, lane = tid & 31;
  int m0 = blockIdx.y * 128, n0 = blockIdx.x * 64;
  v8f acc[4] = {};
  for (int k0 = 0; k0 < K; k0 += 32) {
    for (int i = tid; i < 128 * 32; i += 256) {
      int m = i >> 5, k = i & 31;
      float v = (m0 + m < M) ? A[(size_t)(m0 + m) * K + k0 + k] : 0.0f;
      lA[i] = f2bf(v);
    }
    for (int i = tid; i < 64 * 32; i += 256) {
      int n = i >> 5, k = i & 31;
      lB[i] = f2bf(W[(size_t)(k0 + k) * N + n0 + n]);
    }
    __syncthreads();
    v16bf af = load_a_frag(lA, wave * 16, 0, 32);
#pragma unroll
    for (int t = 0; t < 4; ++t) {
      v16bf bfr = load_b_frag(lB, t * 16, 0, 32);
      acc[t] = __builtin_amdgcn_wmma_f32_16x16x32_bf16(
          false, af, false, bfr, (short)0, acc[t], false, false);
    }
    __syncthreads();
  }
  int mrow = m0 + wave * 16 + ((lane >> 4) << 3);
#pragma unroll
  for (int t = 0; t < 4; ++t) {
    int n = n0 + t * 16 + (lane & 15);
    float bv = bias ? bias[n] : 0.0f;
#pragma unroll
    for (int i = 0; i < 8; ++i) {
      int m = mrow + i;
      if (m < M) {
        float v = acc[t][i] + bv;
        if (ep & 1) v = gelu_tanh(v);
        if (ep & 2) v += R[(size_t)m * N + n];
        C[(size_t)m * N + n] = v;
      }
    }
  }
}

// ---- edge gate: g[E,256](bf16) = (sh@Wa) * silu(sc@Wg) ---------------------
// Block: 128 edges x 64 cols, 8 waves.

__global__ __launch_bounds__(256) void gate_kernel(
    const float* __restrict__ sh, const float* __restrict__ sc,
    const float* __restrict__ Wa, const float* __restrict__ Wg,
    unsigned short* __restrict__ g, int E)
{
  __shared__ unsigned short lSh[128 * 32];  // K=16 zero-padded to 32
  __shared__ unsigned short lSc[128 * 64];
  __shared__ unsigned short lWa[64 * 32];   // [n][k], K padded
  __shared__ unsigned short lWg[64 * 64];   // [n][k]
  int tid = threadIdx.x, wave = tid >> 5, lane = tid & 31;
  int e0 = blockIdx.y * 128, n0 = blockIdx.x * 64;
  for (int i = tid; i < 128 * 32; i += 256) {
    int m = i >> 5, k = i & 31;
    float v = (k < 16 && e0 + m < E) ? sh[(size_t)(e0 + m) * 16 + k] : 0.0f;
    lSh[i] = f2bf(v);
  }
  for (int i = tid; i < 128 * 64; i += 256) {
    int m = i >> 6, k = i & 63;
    float v = (e0 + m < E) ? sc[(size_t)(e0 + m) * 64 + k] : 0.0f;
    lSc[i] = f2bf(v);
  }
  for (int i = tid; i < 64 * 32; i += 256) {
    int n = i >> 5, k = i & 31;
    lWa[i] = f2bf((k < 16) ? Wa[(size_t)k * 256 + n0 + n] : 0.0f);
  }
  for (int i = tid; i < 64 * 64; i += 256) {
    int n = i >> 6, k = i & 63;
    lWg[i] = f2bf(Wg[(size_t)k * 256 + n0 + n]);
  }
  __syncthreads();
  v16bf aSh  = load_a_frag(lSh, wave * 16, 0, 32);
  v16bf aSc0 = load_a_frag(lSc, wave * 16, 0, 64);
  v16bf aSc1 = load_a_frag(lSc, wave * 16, 32, 64);
#pragma unroll
  for (int t = 0; t < 4; ++t) {
    v8f s1 = {}, s2 = {};
    s1 = __builtin_amdgcn_wmma_f32_16x16x32_bf16(false, aSh, false,
          load_b_frag(lWa, t * 16, 0, 32), (short)0, s1, false, false);
    s2 = __builtin_amdgcn_wmma_f32_16x16x32_bf16(false, aSc0, false,
          load_b_frag(lWg, t * 16, 0, 64), (short)0, s2, false, false);
    s2 = __builtin_amdgcn_wmma_f32_16x16x32_bf16(false, aSc1, false,
          load_b_frag(lWg, t * 16, 32, 64), (short)0, s2, false, false);
    int n = n0 + t * 16 + (lane & 15);
    int mb = e0 + wave * 16 + ((lane >> 4) << 3);
#pragma unroll
    for (int i = 0; i < 8; ++i) {
      int e = mb + i;
      if (e < E) {
        float x2 = s2[i];
        float gv = s1[i] * (x2 / (1.0f + expf(-x2)));   // silu
        g[(size_t)e * 256 + n] = f2bf(gv);
      }
    }
  }
}

// ---- edge pass A: logits + segment max (one wave per edge) -----------------

__global__ __launch_bounds__(256) void edge_pass_a(
    const float* __restrict__ Xs, const float* __restrict__ Xd,
    const int* __restrict__ src, const int* __restrict__ dst,
    const unsigned short* __restrict__ gate, const float* __restrict__ gate_row,
    const float* __restrict__ avec, float* __restrict__ s_out,
    float* __restrict__ mx, int E)
{
  int wave = threadIdx.x >> 5, lane = threadIdx.x & 31;
  int e = blockIdx.x * 8 + wave;
  if (e >= E) return;
  int si = src[e], di = dst[e];
  const float* ps = Xs + (size_t)si * 256 + lane * 8;
  const float* pd = Xd + (size_t)di * 256 + lane * 8;
  float m[8];
  if (gate) {
    const unsigned short* pg = gate + (size_t)e * 256 + lane * 8;
#pragma unroll
    for (int j = 0; j < 8; ++j) m[j] = (ps[j] + pd[j]) * bf2f(pg[j]);
  } else {
    const float* pg = gate_row + lane * 8;
#pragma unroll
    for (int j = 0; j < 8; ++j) m[j] = (ps[j] + pd[j]) * pg[j];
  }
  int head = lane >> 2;
  float part = 0.0f;
#pragma unroll
  for (int j = 0; j < 8; ++j) {
    float lr = (m[j] > 0.0f) ? m[j] : 0.2f * m[j];      // leaky_relu(0.2)
    part += lr * avec[head * 32 + (lane & 3) * 8 + j];
  }
  part += __shfl_xor(part, 1);
  part += __shfl_xor(part, 2);
  if ((lane & 3) == 0) {
    s_out[(size_t)e * 8 + head] = part;
    atomicMaxF(&mx[(size_t)di * 8 + head], part);
  }
}

// ---- edge pass B: exp + denominator ----------------------------------------

__global__ void edge_pass_b(const int* __restrict__ dst, float* __restrict__ s_buf,
                            const float* __restrict__ mx, float* __restrict__ den, int E)
{
  int i = blockIdx.x * blockDim.x + threadIdx.x;
  if (i >= E * 8) return;
  int e = i >> 3, h = i & 7;
  int di = dst[e];
  float ex = expf(s_buf[i] - mx[(size_t)di * 8 + h]);
  s_buf[i] = ex;
  atomicAdd(&den[(size_t)di * 8 + h], ex);
}

// ---- edge pass C: alpha-weighted scatter-add -------------------------------

__global__ __launch_bounds__(256) void edge_pass_c(
    const float* __restrict__ Xs, const float* __restrict__ Xd,
    const int* __restrict__ src, const int* __restrict__ dst,
    const unsigned short* __restrict__ gate, const float* __restrict__ gate_row,
    const float* __restrict__ s_buf, const float* __restrict__ den,
    float* __restrict__ agg, int E)
{
  int wave = threadIdx.x >> 5, lane = threadIdx.x & 31;
  int e = blockIdx.x * 8 + wave;
  if (e >= E) return;
  int si = src[e], di = dst[e];
  float al = 0.0f;
  if (lane < 8) al = s_buf[(size_t)e * 8 + lane] / (den[(size_t)di * 8 + lane] + 1e-9f);
  float alpha = __shfl(al, lane >> 2);
  const float* ps = Xs + (size_t)si * 256 + lane * 8;
  const float* pd = Xd + (size_t)di * 256 + lane * 8;
  float m[8];
  if (gate) {
    const unsigned short* pg = gate + (size_t)e * 256 + lane * 8;
#pragma unroll
    for (int j = 0; j < 8; ++j) m[j] = (ps[j] + pd[j]) * bf2f(pg[j]);
  } else {
    const float* pg = gate_row + lane * 8;
#pragma unroll
    for (int j = 0; j < 8; ++j) m[j] = (ps[j] + pd[j]) * pg[j];
  }
  float* pa = agg + (size_t)di * 256 + lane * 8;
#pragma unroll
  for (int j = 0; j < 8; ++j) atomicAdd(&pa[j], m[j] * alpha);
}

// ---- LayerNorm (one wave per row, D=256) -----------------------------------

__global__ __launch_bounds__(256) void ln_kernel(
    const float* __restrict__ X, const float* __restrict__ g,
    const float* __restrict__ b, float* __restrict__ Y, int M)
{
  int wave = threadIdx.x >> 5, lane = threadIdx.x & 31;
  int row = blockIdx.x * 8 + wave;
  if (row >= M) return;
  const float* x = X + (size_t)row * 256 + lane * 8;
  float v[8], s = 0.0f;
#pragma unroll
  for (int j = 0; j < 8; ++j) { v[j] = x[j]; s += v[j]; }
  for (int o = 16; o; o >>= 1) s += __shfl_xor(s, o);
  float mu = s * (1.0f / 256.0f);
  float q = 0.0f;
#pragma unroll
  for (int j = 0; j < 8; ++j) { v[j] -= mu; q += v[j] * v[j]; }
  for (int o = 16; o; o >>= 1) q += __shfl_xor(q, o);
  float rs = rsqrtf(q * (1.0f / 256.0f) + 1e-5f);
  float* y = Y + (size_t)row * 256 + lane * 8;
#pragma unroll
  for (int j = 0; j < 8; ++j) y[j] = v[j] * rs * g[lane * 8 + j] + b[lane * 8 + j];
}

__global__ void fill_kernel(float* __restrict__ p, float v, int n) {
  int i = blockIdx.x * blockDim.x + threadIdx.x;
  if (i < n) p[i] = v;
}

// ---- param offsets (jax pytree = alphabetical dict flattening) -------------
//   e2v:{Wa,Wd,Wo,Ws,a} ffn_he:{W1,W2,b1,b2} ffn_node:{...}
//   ln_he1:{b,g} ln_he2 ln_n1 ln_n2  v2e:{Wa,Wd,Wg,Wo,Ws,a}
enum : size_t {
  P_E2V_WA = 0,        P_E2V_WD = 256,      P_E2V_WO = 65792,
  P_E2V_WS = 131328,   P_E2V_A  = 196864,
  P_FHE_W1 = 197120,   P_FHE_W2 = 328192,   P_FHE_B1 = 459264,  P_FHE_B2 = 459776,
  P_FND_W1 = 460032,   P_FND_W2 = 591104,   P_FND_B1 = 722176,  P_FND_B2 = 722688,
  P_LNHE1_B = 722944,  P_LNHE1_G = 723200,
  P_LNHE2_B = 723456,  P_LNHE2_G = 723712,
  P_LNN1_B  = 723968,  P_LNN1_G  = 724224,
  P_LNN2_B  = 724480,  P_LNN2_G  = 724736,
  P_V2E_WA = 724992,   P_V2E_WD = 729088,   P_V2E_WG = 794624,
  P_V2E_WO = 811008,   P_V2E_WS = 876544,   P_V2E_A  = 942080
};

static inline void launch_gemm(const float* A, const float* W, const float* bias,
                               const float* R, float* C, int M, int N, int K,
                               int ep, hipStream_t s) {
  dim3 grid((N + 63) / 64, (M + 127) / 128);
  gemm_kernel<<<grid, 256, 0, s>>>(A, W, bias, R, C, M, N, K, ep);
}

extern "C" void kernel_launch(void* const* d_in, const int* in_sizes, int n_in,
                              void* d_out, int out_size, void* d_ws, size_t ws_size,
                              hipStream_t stream) {
  const float* node_in  = (const float*)d_in[0];
  const float* he_in    = (const float*)d_in[1];
  const float* sc_nd_in = (const float*)d_in[2];
  const float* sc_he_in = (const float*)d_in[3];
  const float* edge_sh  = (const float*)d_in[4];
  const float* edge_sc  = (const float*)d_in[5];
  const float* prm      = (const float*)d_in[6];
  const int*   v2e_idx  = (const int*)d_in[7];
  const int*   e2v_idx  = (const int*)d_in[8];

  const int NN = in_sizes[0] / D_MODEL;     // 50000 nodes
  const int NH = in_sizes[1] / D_MODEL;     // 25000 hyperedges
  const int E  = in_sizes[7] / 2;           // 300000 incidences

  const int* v2e_src = v2e_idx;       const int* v2e_dst = v2e_idx + E;
  const int* e2v_src = e2v_idx;       const int* e2v_dst = e2v_idx + E;

  // outputs: (out_node, out_he, sc_node, sc_he) concatenated
  float* out_node = (float*)d_out;
  float* out_he   = out_node + (size_t)NN * D_MODEL;
  float* sc_node  = out_he   + (size_t)NH * D_MODEL;
  float* sc_he    = sc_node  + (size_t)NN * D_MODEL;

  // workspace carve-out (~420 MB)
  float* w     = (float*)d_ws;
  float* XsW   = w;                               w += (size_t)NN * D_MODEL;
  float* XdW   = w;                               w += (size_t)NN * D_MODEL;
  float* agg   = w;                               w += (size_t)NN * D_MODEL;
  float* xres  = w;                               w += (size_t)NN * D_MODEL;
  float* lnbuf = w;                               w += (size_t)NN * D_MODEL;
  float* sbuf  = w;                               w += (size_t)E * N_HEADS;
  float* mx    = w;                               w += (size_t)NN * N_HEADS;
  float* den   = w;                               w += (size_t)NN * N_HEADS;
  float* midgate = w;   // union: gate[E,256] bf16  /  mid[M,512] fp32
  unsigned short* gate = (unsigned short*)midgate;
  float* mid = midgate;

  const float NEG_INF = -__builtin_inff();
  dim3 edgeGrid((E + 7) / 8);

  // ================= phase 1: v2e (nodes -> hyperedges) =================
  launch_gemm(node_in, prm + P_V2E_WS, nullptr, nullptr, XsW, NN, 256, 256, 0, stream);
  launch_gemm(he_in,   prm + P_V2E_WD, nullptr, nullptr, XdW, NH, 256, 256, 0, stream);
  { dim3 g(4, (E + 127) / 128);
    gate_kernel<<<g, 256, 0, stream>>>(edge_sh, edge_sc,
        prm + P_V2E_WA, prm + P_V2E_WG, gate, E); }
  fill_kernel<<<(NH * 8 + 255) / 256, 256, 0, stream>>>(mx, NEG_INF, NH * 8);
  fill_kernel<<<(NH * 8 + 255) / 256, 256, 0, stream>>>(den, 0.0f, NH * 8);
  fill_kernel<<<(NH * 256 + 255) / 256, 256, 0, stream>>>(agg, 0.0f, NH * 256);
  edge_pass_a<<<edgeGrid, 256, 0, stream>>>(XsW, XdW, v2e_src, v2e_dst,
      gate, nullptr, prm + P_V2E_A, sbuf, mx, E);
  edge_pass_b<<<(E * 8 + 255) / 256, 256, 0, stream>>>(v2e_dst, sbuf, mx, den, E);
  edge_pass_c<<<edgeGrid, 256, 0, stream>>>(XsW, XdW, v2e_src, v2e_dst,
      gate, nullptr, sbuf, den, agg, E);
  // out_he_pre = shortcut_he + agg @ Wo
  launch_gemm(agg, prm + P_V2E_WO, nullptr, sc_he_in, xres, NH, 256, 256, 2, stream);
  // FFN + residual
  ln_kernel<<<(NH + 7) / 8, 256, 0, stream>>>(xres, prm + P_LNHE1_G, prm + P_LNHE1_B, lnbuf, NH);
  launch_gemm(lnbuf, prm + P_FHE_W1, prm + P_FHE_B1, nullptr, mid, NH, 512, 256, 1, stream);
  launch_gemm(mid,   prm + P_FHE_W2, prm + P_FHE_B2, xres, sc_he, NH, 256, 512, 2, stream);
  ln_kernel<<<(NH + 7) / 8, 256, 0, stream>>>(sc_he, prm + P_LNHE2_G, prm + P_LNHE2_B, out_he, NH);

  // ================= phase 2: e2v (hyperedges -> nodes) =================
  launch_gemm(out_he,  prm + P_E2V_WS, nullptr, nullptr, XsW, NH, 256, 256, 0, stream);
  launch_gemm(node_in, prm + P_E2V_WD, nullptr, nullptr, XdW, NN, 256, 256, 0, stream);
  fill_kernel<<<(NN * 8 + 255) / 256, 256, 0, stream>>>(mx, NEG_INF, NN * 8);
  fill_kernel<<<(NN * 8 + 255) / 256, 256, 0, stream>>>(den, 0.0f, NN * 8);
  fill_kernel<<<(NN * 256 + 255) / 256, 256, 0, stream>>>(agg, 0.0f, NN * 256);
  // gate = ones @ e2v.Wa  ->  constant row = Wa[0,:]
  edge_pass_a<<<edgeGrid, 256, 0, stream>>>(XsW, XdW, e2v_src, e2v_dst,
      nullptr, prm + P_E2V_WA, prm + P_E2V_A, sbuf, mx, E);
  edge_pass_b<<<(E * 8 + 255) / 256, 256, 0, stream>>>(e2v_dst, sbuf, mx, den, E);
  edge_pass_c<<<edgeGrid, 256, 0, stream>>>(XsW, XdW, e2v_src, e2v_dst,
      nullptr, prm + P_E2V_WA, sbuf, den, agg, E);
  launch_gemm(agg, prm + P_E2V_WO, nullptr, sc_nd_in, xres, NN, 256, 256, 2, stream);
  ln_kernel<<<(NN + 7) / 8, 256, 0, stream>>>(xres, prm + P_LNN1_G, prm + P_LNN1_B, lnbuf, NN);
  launch_gemm(lnbuf, prm + P_FND_W1, prm + P_FND_B1, nullptr, mid, NN, 512, 256, 1, stream);
  launch_gemm(mid,   prm + P_FND_W2, prm + P_FND_B2, xres, sc_node, NN, 256, 512, 2, stream);
  ln_kernel<<<(NN + 7) / 8, 256, 0, stream>>>(sc_node, prm + P_LNN2_G, prm + P_LNN2_B, out_node, NN);
}
